// Similariy_Net_75797582840045
// MI455X (gfx1250) — matrix operand order repaired and checked by reading, not compiled, not run on hardware
//
#include <hip/hip_runtime.h>
#include <hip/hip_bf16.h>

// ---------------------------------------------------------------------------
// MI455X (gfx1250) implementation.
// One workgroup (256 threads = 8 wave32) per imputation row n; the 16 peers
// form the M=16 dimension of v_wmma_f32_16x16x32_bf16 tiles.
//  - Weights pre-converted to bf16 in a fragment-interleaved layout so each
//    wave's B operand is one contiguous, fully-coalesced 1KB block.
//  - LDS A/stage buffers use padded strides (bank-conflict-free ds_read_b128).
//  - f32 accumulation, exact-erf GELU on VALU, wave32 shfl reductions.
// ---------------------------------------------------------------------------

typedef __attribute__((ext_vector_type(16))) __bf16 v16bf;
typedef __attribute__((ext_vector_type(8)))  __bf16 v8bf;
typedef __attribute__((ext_vector_type(8)))  float  v8f;

#define F_DIM 128
#define H_DIM 256
#define P_DIM 16

// padded LDS strides (elements)
#define AS_F 136   // bf16 A stride for K=128  (=128+8 -> bank offset 4*l)
#define AS_H 264   // bf16 A stride for K=256  (=256+8)
#define OS_F 132   // f32 stage stride for W=128 (=128+4)
#define OS_H 260   // f32 stage stride for W=256 (=256+4)

__device__ __forceinline__ float gelu_exact(float x) {
    return 0.5f * x * (1.0f + erff(x * 0.70710678118654752f));
}

__device__ __forceinline__ float wred_max(float v) {
#pragma unroll
    for (int o = 16; o > 0; o >>= 1) v = fmaxf(v, __shfl_xor(v, o, 32));
    return v;
}
__device__ __forceinline__ float wred_sum(float v) {
#pragma unroll
    for (int o = 16; o > 0; o >>= 1) v += __shfl_xor(v, o, 32);
    return v;
}

// D(16 x W) = act( A(16 x K) @ W + bias )
// sA: bf16, row stride `astride`.  WT: bf16 fragment-interleaved:
//   [(t*(K/32)+kb)*32 + lane]*16 + e  ->  W[kb*32 + (lane>>4)*16 + e][t*16 + (lane&15)]
// 8 waves cooperate; wave w owns output column tiles w, w+8, ...
__device__ __forceinline__ void gemm16(const __bf16* __restrict__ sA, int astride, int K,
                                       const __bf16* __restrict__ WT,
                                       const float* __restrict__ bias,
                                       float* __restrict__ sD, int dstride, int W,
                                       int wave, int lane, bool do_gelu) {
    const int l  = lane & 15;
    const int hi = lane >> 4;
    const int ntiles = W >> 4;
    const int nkb    = K >> 5;
    for (int t = wave; t < ntiles; t += 8) {
        v8f acc = {};
#pragma unroll
        for (int kb = 0; kb < 8; ++kb) {
            if (kb >= nkb) break;
            const int k0 = kb << 5;
            // A fragment: row l; elems 0..15 = K(k0+16*hi ..)
            const __bf16* ap = sA + l * astride + k0 + hi * 16;
            v8bf a0 = *(const v8bf*)ap;
            v8bf a1 = *(const v8bf*)(ap + 8);
            v16bf av;
#pragma unroll
            for (int i = 0; i < 8; ++i) { av[i] = a0[i]; av[8 + i] = a1[i]; }
            // B fragment: contiguous 32B per lane, 1KB per wave-tile-kblock
            const __bf16* bp = WT + ((size_t)(t * nkb + kb) * 32 + lane) * 16;
            v8bf b0 = *(const v8bf*)bp;
            v8bf b1 = *(const v8bf*)(bp + 8);
            v16bf bv;
#pragma unroll
            for (int i = 0; i < 8; ++i) { bv[i] = b0[i]; bv[8 + i] = b1[i]; }
            acc = __builtin_amdgcn_wmma_f32_16x16x32_bf16(false, av, false, bv,
                                                          (short)0, acc, false, false);
        }
        const int c0 = t << 4;
        const float bb = bias ? bias[c0 + l] : 0.0f;
#pragma unroll
        for (int r = 0; r < 8; ++r) {
            float x = acc[r] + bb;                 // row M = r + 8*hi, col N = c0+l
            if (do_gelu) x = gelu_exact(x);
            sD[(r + 8 * hi) * dstride + c0 + l] = x;
        }
    }
}

// ---------------------------------------------------------------------------
// Prep kernel: convert f32 weight (K x C) to bf16.
// mode 1: fragment-interleaved layout for gemm16 (see above)
// mode 0: plain transposed (C x K) layout (used for sc_w thread-dot)
// ---------------------------------------------------------------------------
__global__ void wt_prep_bf16(const float* __restrict__ src,
                             __bf16* __restrict__ dst, int K, int C, int mode) {
    int idx = blockIdx.x * blockDim.x + threadIdx.x;
    int total = K * C;
    if (idx >= total) return;
    if (mode == 1) {
        int e    = idx & 15;
        int lane = (idx >> 4) & 31;
        int rest = idx >> 9;
        int nkb  = K >> 5;
        int kb   = rest % nkb;
        int t    = rest / nkb;
        int col  = t * 16 + (lane & 15);
        int k    = kb * 32 + (lane >> 4) * 16 + e;
        dst[idx] = (__bf16)src[k * C + col];
    } else {
        int c = idx / K;
        int k = idx - c * K;
        dst[idx] = (__bf16)src[k * C + c];
    }
}

// ---------------------------------------------------------------------------
// Prep kernel: G = FEA @ FEA^T  (128x128, K=256) with WMMA. 64 blocks x 1 wave.
// ---------------------------------------------------------------------------
__global__ __launch_bounds__(32) void gram_wmma(const float* __restrict__ FEA,
                                                float* __restrict__ G) {
    const int ti = (blockIdx.x & 7) << 4;
    const int tj = (blockIdx.x >> 3) << 4;
    const int lane = threadIdx.x;
    const int l  = lane & 15;
    const int hi = lane >> 4;
    v8f acc = {};
#pragma unroll
    for (int kb = 0; kb < 8; ++kb) {
        const int k0 = kb << 5;
        const float* ap = FEA + (size_t)(ti + l) * H_DIM + k0 + hi * 8;
        const float* bp = FEA + (size_t)(tj + l) * H_DIM + k0 + hi * 8;
        v16bf av, bv;
#pragma unroll
        for (int i = 0; i < 8; ++i) {
            av[i]     = (__bf16)ap[i];
            av[8 + i] = (__bf16)ap[16 + i];
            bv[i]     = (__bf16)bp[i];
            bv[8 + i] = (__bf16)bp[16 + i];
        }
        acc = __builtin_amdgcn_wmma_f32_16x16x32_bf16(false, av, false, bv,
                                                      (short)0, acc, false, false);
    }
#pragma unroll
    for (int r = 0; r < 8; ++r)
        G[(size_t)(ti + r + 8 * hi) * F_DIM + tj + l] = acc[r];
}

// ---------------------------------------------------------------------------
// Main kernel: one block per n
// ---------------------------------------------------------------------------
struct Params {
    const float* Mmat;      // (R,128)
    const float* OBS;       // (R,256)
    const float* G;         // (128,128) f32
    const int*   IMP_OBS;   // (N,)
    const int*   IMP_FEA;   // (N,)
    const int*   PEER;      // (N,16)
    // fragment-interleaved bf16 weights
    const __bf16 *w_irm1, *w_irm2, *w_irr, *w_irc;
    const __bf16 *w_prm1, *w_prm2, *w_prr, *w_prc;
    const __bf16 *w_sr1, *w_sr2, *w_sm1, *w_sm2;
    const __bf16 *w_sc;     // plain transposed (256x256)
    // f32 biases
    const float *b_irm1, *b_irm2, *b_irr, *b_irc;
    const float *b_prm1, *b_prm2, *b_prr, *b_prc;
    const float *b_sr1, *b_sr2, *b_sm1, *b_sm2, *b_sc;
    float* out_csim;        // (N,256)
    float* out_sim;         // (N,16)
};

__global__ __launch_bounds__(256) void sim_main(Params P) {
    __shared__ __bf16 sA[P_DIM * AS_H];        // staged WMMA A operand (padded)
    __shared__ float  sOut[P_DIM * OS_H];      // stage output (padded)
    __shared__ float  sCi[P_DIM * OS_H];       // c_i (padded)
    __shared__ float  sObsP[P_DIM * H_DIM];    // OBS_embs[peers]
    __shared__ float  sMpeer[P_DIM * F_DIM];   // M[peers]
    __shared__ float  sARow[F_DIM];            // a_ji row of Gram
    __shared__ float  sObsI[H_DIM];            // OBS_embs[imp]
    __shared__ float  sMimp[F_DIM];            // M[imp]
    __shared__ float  sSim[P_DIM];
    __shared__ float  sEsim[H_DIM];
    __shared__ int    sPeer[P_DIM];
    __shared__ int    sIdx[2];

    const int n    = blockIdx.x;
    const int tid  = threadIdx.x;
    const int wave = tid >> 5;
    const int lane = tid & 31;

    if (tid == 0) { sIdx[0] = P.IMP_FEA[n]; sIdx[1] = P.IMP_OBS[n]; }
    if (tid < P_DIM) sPeer[tid] = P.PEER[n * P_DIM + tid];
    __syncthreads();
    const int f = sIdx[0];
    const int o = sIdx[1];

    // ---- gathers ----
    for (int i = tid; i < F_DIM; i += 256) {
        sARow[i] = P.G[(size_t)f * F_DIM + i];
        sMimp[i] = P.Mmat[(size_t)o * F_DIM + i];
    }
    for (int i = tid; i < H_DIM; i += 256) sObsI[i] = P.OBS[(size_t)o * H_DIM + i];
    for (int i = tid; i < P_DIM * F_DIM; i += 256) {
        int p = i >> 7, j = i & (F_DIM - 1);
        sMpeer[i] = P.Mmat[(size_t)sPeer[p] * F_DIM + j];
    }
    for (int i = tid; i < P_DIM * H_DIM; i += 256) {
        int p = i >> 8, j = i & (H_DIM - 1);
        sObsP[i] = P.OBS[(size_t)sPeer[p] * H_DIM + j];
    }
    __syncthreads();

    // =================== c_i chain (mask varies per peer) ===================
    for (int p = wave; p < P_DIM; p += 8) {
        float v[4], mx = -1e30f;
#pragma unroll
        for (int q = 0; q < 4; ++q) {
            int j = lane + q * 32;
            float z = sMpeer[p * F_DIM + j] * ((j == f) ? 0.0f : 1.0f);
            v[q] = z; mx = fmaxf(mx, z);
        }
        mx = wred_max(mx);
        float sum = 0.0f;
#pragma unroll
        for (int q = 0; q < 4; ++q) { v[q] = __expf(v[q] - mx); sum += v[q]; }
        sum = wred_sum(sum);
        float inv = 1.0f / sum;
#pragma unroll
        for (int q = 0; q < 4; ++q) sA[p * AS_F + lane + q * 32] = (__bf16)(v[q] * inv);
    }
    __syncthreads();
    gemm16(sA, AS_F, F_DIM, P.w_irm1, P.b_irm1, sOut, OS_H, H_DIM, wave, lane, true);
    __syncthreads();
    for (int i = tid; i < P_DIM * H_DIM; i += 256) {
        int p = i >> 8, j = i & (H_DIM - 1);
        sA[p * AS_H + j] = (__bf16)sOut[p * OS_H + j];
    }
    __syncthreads();
    gemm16(sA, AS_H, H_DIM, P.w_irm2, P.b_irm2, sOut, OS_F, F_DIM, wave, lane, true);
    __syncthreads();
    for (int i = tid; i < P_DIM * F_DIM; i += 256) {
        int p = i >> 7, j = i & (F_DIM - 1);
        sA[p * AS_F + j] = (__bf16)(sARow[j] * sOut[p * OS_F + j]);
    }
    __syncthreads();
    gemm16(sA, AS_F, F_DIM, P.w_irr, P.b_irr, sOut, OS_H, H_DIM, wave, lane, true);
    __syncthreads();
    for (int i = tid; i < P_DIM * H_DIM; i += 256) {
        int p = i >> 8, j = i & (H_DIM - 1);
        sA[p * AS_H + j] = (__bf16)(sObsI[j] * sOut[p * OS_H + j]);
    }
    __syncthreads();
    gemm16(sA, AS_H, H_DIM, P.w_irc, P.b_irc, sCi, OS_H, H_DIM, wave, lane, true);
    __syncthreads();

    // =================== c_t chain (mask fixed per n, rows replicated) ======
    for (int p = wave; p < P_DIM; p += 8) {
        float v[4], mx = -1e30f;
#pragma unroll
        for (int q = 0; q < 4; ++q) {
            int j = lane + q * 32;
            float z = sMimp[j] * ((j == f) ? 0.0f : 1.0f);
            v[q] = z; mx = fmaxf(mx, z);
        }
        mx = wred_max(mx);
        float sum = 0.0f;
#pragma unroll
        for (int q = 0; q < 4; ++q) { v[q] = __expf(v[q] - mx); sum += v[q]; }
        sum = wred_sum(sum);
        float inv = 1.0f / sum;
#pragma unroll
        for (int q = 0; q < 4; ++q) sA[p * AS_F + lane + q * 32] = (__bf16)(v[q] * inv);
    }
    __syncthreads();
    gemm16(sA, AS_F, F_DIM, P.w_prm1, P.b_prm1, sOut, OS_H, H_DIM, wave, lane, true);
    __syncthreads();
    for (int i = tid; i < P_DIM * H_DIM; i += 256) {
        int p = i >> 8, j = i & (H_DIM - 1);
        sA[p * AS_H + j] = (__bf16)sOut[p * OS_H + j];
    }
    __syncthreads();
    gemm16(sA, AS_H, H_DIM, P.w_prm2, P.b_prm2, sOut, OS_F, F_DIM, wave, lane, true);
    __syncthreads();
    for (int i = tid; i < P_DIM * F_DIM; i += 256) {
        int p = i >> 7, j = i & (F_DIM - 1);
        sA[p * AS_F + j] = (__bf16)(sARow[j] * sOut[p * OS_F + j]);
    }
    __syncthreads();
    gemm16(sA, AS_F, F_DIM, P.w_prr, P.b_prr, sOut, OS_H, H_DIM, wave, lane, true);
    __syncthreads();
    for (int i = tid; i < P_DIM * H_DIM; i += 256) {
        int p = i >> 8, j = i & (H_DIM - 1);
        sA[p * AS_H + j] = (__bf16)(sObsP[i] * sOut[p * OS_H + j]);
    }
    __syncthreads();
    gemm16(sA, AS_H, H_DIM, P.w_prc, P.b_prc, sOut, OS_H, H_DIM, wave, lane, true);  // c_t
    __syncthreads();

    // sim[p] = <c_i[p], c_t[p]>
    for (int p = wave; p < P_DIM; p += 8) {
        float s = 0.0f;
#pragma unroll
        for (int q = 0; q < 8; ++q) {
            int j = lane + q * 32;
            s += sCi[p * OS_H + j] * sOut[p * OS_H + j];
        }
        s = wred_sum(s);
        if (lane == 0) sSim[p] = s;
    }
    __syncthreads();

    // =================== phi_sr / phi_sm ===================
    // m_q = [m_peer | onehot(f)]  (16x256)
    for (int i = tid; i < P_DIM * H_DIM; i += 256) {
        int p = i >> 8, j = i & (H_DIM - 1);
        float v = (j < F_DIM) ? sMpeer[p * F_DIM + j] : ((j - F_DIM == f) ? 1.0f : 0.0f);
        sA[p * AS_H + j] = (__bf16)v;
    }
    __syncthreads();
    gemm16(sA, AS_H, H_DIM, P.w_sr1, P.b_sr1, sOut, OS_H, H_DIM, wave, lane, true);
    __syncthreads();
    for (int i = tid; i < P_DIM * H_DIM; i += 256) {
        int p = i >> 8, j = i & (H_DIM - 1);
        sA[p * AS_H + j] = (__bf16)sOut[p * OS_H + j];
    }
    __syncthreads();
    gemm16(sA, AS_H, H_DIM, P.w_sr2, P.b_sr2, sOut, OS_H, H_DIM, wave, lane, true);  // r_t_j
    __syncthreads();
    for (int i = tid; i < P_DIM * H_DIM; i += 256) {
        int p = i >> 8, j = i & (H_DIM - 1);
        sA[p * AS_H + j] = (__bf16)(sObsP[i] * sOut[p * OS_H + j]);
    }
    __syncthreads();
    gemm16(sA, AS_H, H_DIM, P.w_sm1, P.b_sm1, sOut, OS_H, H_DIM, wave, lane, true);
    __syncthreads();
    for (int i = tid; i < P_DIM * H_DIM; i += 256) {
        int p = i >> 8, j = i & (H_DIM - 1);
        sA[p * AS_H + j] = (__bf16)sOut[p * OS_H + j];
    }
    __syncthreads();
    gemm16(sA, AS_H, H_DIM, P.w_sm2, P.b_sm2, sOut, OS_H, H_DIM, wave, lane, true);  // h_query
    __syncthreads();

    // e_sim[h] = sum_p sim[p] * h_query[p][h]
    {
        float e = 0.0f;
#pragma unroll
        for (int p = 0; p < P_DIM; ++p) e += sSim[p] * sOut[p * OS_H + tid];
        sEsim[tid] = e;
    }
    __syncthreads();

    // c_sim[h] = gelu( e_sim @ sc_w + b ) — one output per thread, plain WT row
    {
        const __bf16* wr = P.w_sc + (size_t)tid * H_DIM;
        float acc = P.b_sc[tid];
        for (int k = 0; k < H_DIM; ++k) acc += sEsim[k] * (float)wr[k];
        P.out_csim[(size_t)n * H_DIM + tid] = gelu_exact(acc);
    }
    if (tid < P_DIM) P.out_sim[(size_t)n * P_DIM + tid] = sSim[tid];
}

// ---------------------------------------------------------------------------
extern "C" void kernel_launch(void* const* d_in, const int* in_sizes, int n_in,
                              void* d_out, int out_size, void* d_ws, size_t ws_size,
                              hipStream_t stream) {
    const float* Mmat = (const float*)d_in[0];
    const float* OBS  = (const float*)d_in[1];
    const float* FEA  = (const float*)d_in[2];
    const int* IMP_OBS = (const int*)d_in[3];
    const int* IMP_FEA = (const int*)d_in[4];
    const int* PEER    = (const int*)d_in[5];
    const int N = in_sizes[3];

    // workspace layout: G (128*128 f32) then bf16 weights
    float* G = (float*)d_ws;
    __bf16* wbase = (__bf16*)((char*)d_ws + (size_t)F_DIM * F_DIM * sizeof(float));
    size_t woff = 0;

    struct WDesc { int idx, K, C, mode; };
    const WDesc wd[13] = {
        {6,  F_DIM, H_DIM, 1},     // i_rm_w1
        {8,  H_DIM, F_DIM, 1},     // i_rm_w2
        {10, F_DIM, H_DIM, 1},     // i_rr_w
        {12, H_DIM, H_DIM, 1},     // i_rc_w
        {14, F_DIM, H_DIM, 1},     // p_rm_w1
        {16, H_DIM, F_DIM, 1},     // p_rm_w2
        {18, F_DIM, H_DIM, 1},     // p_rr_w
        {20, H_DIM, H_DIM, 1},     // p_rc_w
        {22, 2 * F_DIM, H_DIM, 1}, // sr_w1
        {24, H_DIM, H_DIM, 1},     // sr_w2
        {26, H_DIM, H_DIM, 1},     // sm_w1
        {28, H_DIM, H_DIM, 1},     // sm_w2
        {30, H_DIM, H_DIM, 0},     // sc_w (plain transposed)
    };
    const __bf16* wt[13];
    for (int i = 0; i < 13; ++i) {
        __bf16* dst = wbase + woff;
        int total = wd[i].K * wd[i].C;
        wt_prep_bf16<<<(total + 255) / 256, 256, 0, stream>>>(
            (const float*)d_in[wd[i].idx], dst, wd[i].K, wd[i].C, wd[i].mode);
        wt[i] = dst;
        woff += (size_t)total;
    }

    gram_wmma<<<64, 32, 0, stream>>>(FEA, G);

    Params P;
    P.Mmat = Mmat; P.OBS = OBS; P.G = G;
    P.IMP_OBS = IMP_OBS; P.IMP_FEA = IMP_FEA; P.PEER = PEER;
    P.w_irm1 = wt[0]; P.w_irm2 = wt[1]; P.w_irr = wt[2]; P.w_irc = wt[3];
    P.w_prm1 = wt[4]; P.w_prm2 = wt[5]; P.w_prr = wt[6]; P.w_prc = wt[7];
    P.w_sr1 = wt[8];  P.w_sr2 = wt[9];  P.w_sm1 = wt[10]; P.w_sm2 = wt[11];
    P.w_sc = wt[12];
    P.b_irm1 = (const float*)d_in[7];  P.b_irm2 = (const float*)d_in[9];
    P.b_irr  = (const float*)d_in[11]; P.b_irc  = (const float*)d_in[13];
    P.b_prm1 = (const float*)d_in[15]; P.b_prm2 = (const float*)d_in[17];
    P.b_prr  = (const float*)d_in[19]; P.b_prc  = (const float*)d_in[21];
    P.b_sr1  = (const float*)d_in[23]; P.b_sr2  = (const float*)d_in[25];
    P.b_sm1  = (const float*)d_in[27]; P.b_sm2  = (const float*)d_in[29];
    P.b_sc   = (const float*)d_in[31];
    P.out_csim = (float*)d_out;
    P.out_sim  = (float*)d_out + (size_t)N * H_DIM;

    sim_main<<<N, 256, 0, stream>>>(P);
}
